// FMN_38405597561654
// MI455X (gfx1250) — matrix operand order repaired and checked by reading, not compiled, not run on hardware
//
#include <hip/hip_runtime.h>
#include <hip/hip_bf16.h>

// ---------------------------------------------------------------------------
// Types for CDNA5 WMMA (wave32): V_WMMA_F32_16X16X32_BF16
// ---------------------------------------------------------------------------
typedef __bf16 bf16;
typedef __attribute__((ext_vector_type(16))) __bf16 v16bf;
typedef __attribute__((ext_vector_type(8)))  __bf16 v8bf;
typedef __attribute__((ext_vector_type(8)))  float  v8f;

#define H 256
#define G3 768   // 3*H gate width
#define QB 64    // queries
#define LQ 20
#define DB 12    // docs per set
#define DPAD 16  // padded doc batch (one 16-row M tile)
#define LD 50

__device__ __forceinline__ v8f wmma_bf16(v16bf a, v16bf b, v8f c) {
  // D = A(16x32) * B(32x16) + C ; fp32 accumulate
  return __builtin_amdgcn_wmma_f32_16x16x32_bf16(
      /*neg_a=*/false, a, /*neg_b=*/false, b,
      /*c_mod=*/(short)0, c, /*reuse_a=*/false, /*reuse_b=*/false);
}

__device__ __forceinline__ float sigmoidf_(float x) {
  return 1.0f / (1.0f + __expf(-x));
}

// A-fragment: 16x32 bf16, row-major source with row stride 256 (K=256 always).
// Lane l (laneHi = l>=16): row = base + l%16.
// elements e=0..15 -> K = kt*32 + (laneHi?16:0) + e
__device__ __forceinline__ v16bf load_a_frag(const bf16* __restrict__ A,
                                             int row, int kt, int laneHi) {
  const bf16* p = A + (size_t)row * H + kt * 32 + (laneHi ? 16 : 0);
  v8bf lo = *reinterpret_cast<const v8bf*>(p);
  v8bf hi = *reinterpret_cast<const v8bf*>(p + 8);
  v16bf a;
#pragma unroll
  for (int e = 0; e < 8; ++e) { a[e] = lo[e]; a[e + 8] = hi[e]; }
  return a;
}

// B-fragment: 32x16 bf16 where B[k][n] = W[n][k] (W row-major [768][256]).
// Lane l: n = base_n + l%16 ; elements e=0..15 -> K = kt*32 + (laneHi?16:0) + e
// => one contiguous 32-byte read per lane.
__device__ __forceinline__ v16bf load_b_frag(const bf16* __restrict__ W,
                                             int n, int kt, int laneHi) {
  const bf16* p = W + (size_t)n * H + kt * 32 + (laneHi ? 16 : 0);
  return *reinterpret_cast<const v16bf*>(p);
}

// ---------------------------------------------------------------------------
// Weight fp32 -> bf16 conversion (7 matrices of [768][256], packed contiguous)
// ---------------------------------------------------------------------------
__global__ void convert_weights(const float* w0, const float* w1, const float* w2,
                                const float* w3, const float* w4, const float* w5,
                                const float* w6, bf16* __restrict__ out) {
  const float* srcs[7] = {w0, w1, w2, w3, w4, w5, w6};
  const int per = G3 * H;
  int idx = blockIdx.x * blockDim.x + threadIdx.x;
  int total = 7 * per;
  for (; idx < total; idx += gridDim.x * blockDim.x) {
    int w = idx / per, e = idx - w * per;
    out[idx] = (bf16)srcs[w][e];
  }
}

// ---------------------------------------------------------------------------
// Embedding gather -> time-major bf16 X: row = t*Bpad + b, tokens are [B][L]
// ---------------------------------------------------------------------------
__global__ void gather_embed(const int* __restrict__ tok,
                             const float* __restrict__ emb,
                             bf16* __restrict__ X, int L, int B, int Bpad) {
  int idx = blockIdx.x * blockDim.x + threadIdx.x;
  int total = L * Bpad * H;
  for (; idx < total; idx += gridDim.x * blockDim.x) {
    int row = idx >> 8, k = idx & (H - 1);
    int t = row / Bpad, b = row - t * Bpad;
    float v = 0.0f;
    if (b < B) v = emb[(size_t)tok[b * L + t] * H + k];
    X[idx] = (bf16)v;
  }
}

// ---------------------------------------------------------------------------
// Batched gate GEMM: C[m][n] = sum_k A[m][k]*W[n][k] + bias[n]
// A bf16 [Mrows][256], W bf16 [768][256], C fp32 [Mrows][768]
// grid = (12, Mrows/16), block = 32; each wave computes a 1x4 strip of
// 16x16 tiles so every A-fragment is amortized over 4 WMMAs.
// ---------------------------------------------------------------------------
__global__ void __launch_bounds__(32)
wmma_gemm_bias(const bf16* __restrict__ A, const bf16* __restrict__ W,
               const float* __restrict__ bias, float* __restrict__ C) {
  const int nt0 = blockIdx.x * 4, mt = blockIdx.y;
  const int lane = threadIdx.x & 31;
  const int laneHi = lane >> 4, l16 = lane & 15;
  const int row = mt * 16 + l16;
  v8f acc[4] = {{}, {}, {}, {}};
#pragma unroll
  for (int kt = 0; kt < 8; ++kt) {
    if (kt < 7)  // prefetch next A slab (global_prefetch_b8)
      __builtin_prefetch(A + (size_t)row * H + (kt + 1) * 32, 0, 1);
    v16bf a = load_a_frag(A, row, kt, laneHi);
#pragma unroll
    for (int q = 0; q < 4; ++q) {
      v16bf b = load_b_frag(W, (nt0 + q) * 16 + l16, kt, laneHi);
      acc[q] = wmma_bf16(a, b, acc[q]);
    }
  }
#pragma unroll
  for (int q = 0; q < 4; ++q) {
    const int n = (nt0 + q) * 16 + l16;
    const float bn = bias[n];
#pragma unroll
    for (int r = 0; r < 8; ++r) {
      int m = mt * 16 + r + (laneHi ? 8 : 0);
      C[(size_t)m * G3 + n] = acc[q][r] + bn;
    }
  }
}

// ---------------------------------------------------------------------------
// Persistent query GRU scan: one workgroup, 16 waves, h in LDS (bf16).
// Wave w owns gate-column slice jt = w (16 columns) for ALL 4 M-tiles:
// per kt the 3 B-fragments (r/z/n rows of Whh) are loaded ONCE and reused
// across 4 A-fragments -> 12 WMMAs per 3 global B loads, 4x less L2 traffic
// on the sequential critical path.
// ---------------------------------------------------------------------------
__global__ void __launch_bounds__(512)
gru_scan_query(const float* __restrict__ Gi,   // [20][64][768], bih folded in
               const bf16* __restrict__ Whh,   // [768][256] bf16
               const float* __restrict__ bhh,  // [768]
               float* __restrict__ hOut, bf16* __restrict__ hOutB) {
  __shared__ __align__(16) bf16 hS[QB * H];
  const int tid = threadIdx.x;
  const int wave = tid >> 5, lane = tid & 31;
  const int laneHi = lane >> 4, l16 = lane & 15;
  const int jt = wave;              // 0..15: this wave's gate-column tile
  const int j = jt * 16 + l16;      // gate column in [0,256)

  for (int i = tid; i < QB * H; i += 512) hS[i] = (bf16)0.0f;
  __syncthreads();

  const float br_ = bhh[j], bz_ = bhh[H + j], bn_ = bhh[2 * H + j];

  for (int t = 0; t < LQ; ++t) {
    v8f accR[4] = {{}, {}, {}, {}};
    v8f accZ[4] = {{}, {}, {}, {}};
    v8f accN[4] = {{}, {}, {}, {}};
#pragma unroll
    for (int kt = 0; kt < 8; ++kt) {
      v16bf brf = load_b_frag(Whh, j,         kt, laneHi);
      v16bf bzf = load_b_frag(Whh, H + j,     kt, laneHi);
      v16bf bnf = load_b_frag(Whh, 2 * H + j, kt, laneHi);
#pragma unroll
      for (int mt = 0; mt < 4; ++mt) {
        v16bf a = load_a_frag(hS, mt * 16 + l16, kt, laneHi);  // ds reads
        accR[mt] = wmma_bf16(a, brf, accR[mt]);
        accZ[mt] = wmma_bf16(a, bzf, accZ[mt]);
        accN[mt] = wmma_bf16(a, bnf, accN[mt]);
      }
    }
    __syncthreads();  // everyone done reading h_t from LDS
#pragma unroll
    for (int mt = 0; mt < 4; ++mt) {
#pragma unroll
      for (int rI = 0; rI < 8; ++rI) {
        const int m = mt * 16 + rI + (laneHi ? 8 : 0);
        const float* g = Gi + ((size_t)t * QB + m) * G3;
        float rr = sigmoidf_(g[j]         + accR[mt][rI] + br_);
        float zz = sigmoidf_(g[H + j]     + accZ[mt][rI] + bz_);
        float nn = tanhf(g[2 * H + j] + rr * (accN[mt][rI] + bn_));
        float hOld = (float)hS[m * H + j];
        float hNew = (1.0f - zz) * nn + zz * hOld;
        hS[m * H + j] = (bf16)hNew;
        if (t == LQ - 1) {
          hOut[m * H + j]  = hNew;
          hOutB[m * H + j] = (bf16)hNew;
        }
      }
    }
    __syncthreads();  // h_{t+1} visible before next step's A-frag loads
  }
}

// ---------------------------------------------------------------------------
// Persistent doc scan: two GRUs (content c, attention a) over 50 steps.
// grid=2: block 0 = positive set, block 1 = negative set (run on 2 WGPs
// concurrently). 8 waves; 32 tile-groups = (gru in {c,a}) x (jt in 0..15).
// ---------------------------------------------------------------------------
__global__ void __launch_bounds__(256)
gru_scan_docs(const float* __restrict__ GiPC, const float* __restrict__ GiPA,
              const float* __restrict__ GiNC, const float* __restrict__ GiNA,
              const bf16* __restrict__ WhhC, const bf16* __restrict__ WhhA,
              const float* __restrict__ bhhC, const float* __restrict__ bhhA,
              float* __restrict__ hcP, float* __restrict__ haP,
              float* __restrict__ hcN, float* __restrict__ haN) {
  const int set = blockIdx.x;  // 0 = pos, 1 = neg
  const float* GiC = set ? GiNC : GiPC;
  const float* GiA = set ? GiNA : GiPA;
  float* hcOut = set ? hcN : hcP;
  float* haOut = set ? haN : haP;

  __shared__ __align__(16) bf16 hC[DPAD * H];
  __shared__ __align__(16) bf16 hA[DPAD * H];
  const int tid = threadIdx.x;
  const int wave = tid >> 5, lane = tid & 31;
  const int laneHi = lane >> 4, l16 = lane & 15;

  for (int i = tid; i < DPAD * H; i += 256) { hC[i] = (bf16)0.0f; hA[i] = (bf16)0.0f; }
  __syncthreads();

  int gru[4], jt[4];
#pragma unroll
  for (int i = 0; i < 4; ++i) { int g = wave * 4 + i; gru[i] = g >> 4; jt[i] = g & 15; }

  for (int t = 0; t < LD; ++t) {
    v8f ar[4], az[4], an[4];
#pragma unroll
    for (int i = 0; i < 4; ++i) {
      const bf16* hp = gru[i] ? hA : hC;
      const bf16* W  = gru[i] ? WhhA : WhhC;
      v8f r = {}, z = {}, n = {};
      const int jr = jt[i] * 16 + l16;
#pragma unroll
      for (int kt = 0; kt < 8; ++kt) {
        v16bf a  = load_a_frag(hp, l16, kt, laneHi);
        v16bf br = load_b_frag(W, jr,          kt, laneHi);
        v16bf bz = load_b_frag(W, H + jr,      kt, laneHi);
        v16bf bn = load_b_frag(W, 2 * H + jr,  kt, laneHi);
        r = wmma_bf16(a, br, r);
        z = wmma_bf16(a, bz, z);
        n = wmma_bf16(a, bn, n);
      }
      ar[i] = r; az[i] = z; an[i] = n;
    }
    __syncthreads();
#pragma unroll
    for (int i = 0; i < 4; ++i) {
      const float* Gi  = gru[i] ? GiA : GiC;
      const float* bhh = gru[i] ? bhhA : bhhC;
      bf16* hp   = gru[i] ? hA : hC;
      float* out = gru[i] ? haOut : hcOut;
      const int j = jt[i] * 16 + l16;
      const float br_ = bhh[j], bz_ = bhh[H + j], bn_ = bhh[2 * H + j];
#pragma unroll
      for (int rI = 0; rI < 8; ++rI) {
        const int m = rI + (laneHi ? 8 : 0);
        const float* g = Gi + ((size_t)t * DPAD + m) * G3;
        float rr = sigmoidf_(g[j]         + ar[i][rI] + br_);
        float zz = sigmoidf_(g[H + j]     + az[i][rI] + bz_);
        float nn = tanhf(g[2 * H + j] + rr * (an[i][rI] + bn_));
        float hOld = (float)hp[m * H + j];
        float hNew = (1.0f - zz) * nn + zz * hOld;
        hp[m * H + j] = (bf16)hNew;
        if (t == LD - 1) out[m * H + j] = hNew;
      }
    }
    __syncthreads();
  }
}

// ---------------------------------------------------------------------------
// position_dense: c'[d][o] = bd[o] + hc[d][:256].Wd[o][:256] + pe[d][:4].Wd[o][256:260]
// grid = (12, 2): y selects pos/neg set so both run concurrently.
// ---------------------------------------------------------------------------
__global__ void __launch_bounds__(256)
pos_dense(const float* __restrict__ hcP, const float* __restrict__ hcN,
          const float* __restrict__ posTable,
          const float* __restrict__ Wd, const float* __restrict__ bd,
          float* __restrict__ outP, float* __restrict__ outN) {
  const int d = blockIdx.x;    // 0..11
  const float* hc = blockIdx.y ? hcN : hcP;
  float* outC     = blockIdx.y ? outN : outP;
  const int o = threadIdx.x;   // 0..255
  const float* w = Wd + (size_t)o * (H + 4);
  float acc = bd[o];
  for (int k = 0; k < H; ++k) acc += hc[d * H + k] * w[k];
#pragma unroll
  for (int p = 0; p < 4; ++p) acc += posTable[d * 4 + p] * w[H + p];
  outC[d * H + o] = acc;
}

// ---------------------------------------------------------------------------
// Memory read: q_aug = q + softmax(q.aP^T).cP + softmax(q.aN^T).cN
// one block per query; wave shuffles + LDS atomics for the 24 dot products
// ---------------------------------------------------------------------------
__global__ void __launch_bounds__(256)
attend_combine(const float* __restrict__ q,
               const float* __restrict__ aP, const float* __restrict__ cP,
               const float* __restrict__ aN, const float* __restrict__ cN,
               float* __restrict__ qaug, bf16* __restrict__ qaugB) {
  const int Q = blockIdx.x;
  const int tid = threadIdx.x;
  const int lane = tid & 31;
  __shared__ float sc[2 * DB];
  if (tid < 2 * DB) sc[tid] = 0.0f;
  __syncthreads();

  const float e = q[Q * H + tid];
  for (int d = 0; d < DB; ++d) {
    float p = e * aP[d * H + tid];
    float n = e * aN[d * H + tid];
#pragma unroll
    for (int m = 16; m >= 1; m >>= 1) {
      p += __shfl_xor(p, m, 32);
      n += __shfl_xor(n, m, 32);
    }
    if (lane == 0) { atomicAdd(&sc[d], p); atomicAdd(&sc[DB + d], n); }
  }
  __syncthreads();
  if (tid == 0) {
    for (int g = 0; g < 2; ++g) {
      float mx = -1e30f;
      for (int d = 0; d < DB; ++d) mx = fmaxf(mx, sc[g * DB + d]);
      float s = 0.0f;
      for (int d = 0; d < DB; ++d) { float ev = __expf(sc[g * DB + d] - mx); sc[g * DB + d] = ev; s += ev; }
      float inv = 1.0f / s;
      for (int d = 0; d < DB; ++d) sc[g * DB + d] *= inv;
    }
  }
  __syncthreads();
  float acc = e;
  for (int d = 0; d < DB; ++d)
    acc += sc[d] * cP[d * H + tid] + sc[DB + d] * cN[d * H + tid];
  qaug[Q * H + tid]  = acc;
  qaugB[Q * H + tid] = (bf16)acc;
}

// ---------------------------------------------------------------------------
// Final GRU step with h0 = 0  =>  gh = bhh exactly; out = (1-z)*n
// ---------------------------------------------------------------------------
__global__ void __launch_bounds__(256)
final_gate(const float* __restrict__ Gf, const float* __restrict__ bhh,
           float* __restrict__ out) {
  const int idx = blockIdx.x * 256 + threadIdx.x;  // 64*256
  const int qi = idx >> 8, j = idx & (H - 1);
  const float* g = Gf + (size_t)qi * G3;
  float r = sigmoidf_(g[j] + bhh[j]);
  float z = sigmoidf_(g[H + j] + bhh[H + j]);
  float n = tanhf(g[2 * H + j] + r * bhh[2 * H + j]);
  out[idx] = (1.0f - z) * n;
}

// ---------------------------------------------------------------------------
// Host launch
// ---------------------------------------------------------------------------
extern "C" void kernel_launch(void* const* d_in, const int* in_sizes, int n_in,
                              void* d_out, int out_size, void* d_ws, size_t ws_size,
                              hipStream_t stream) {
  (void)in_sizes; (void)n_in; (void)out_size; (void)ws_size;
  // input order per setup_inputs()
  const int*   query_tokens = (const int*)d_in[0];
  const int*   pos_tokens   = (const int*)d_in[1];
  const int*   neg_tokens   = (const int*)d_in[2];
  const float* term_emb     = (const float*)d_in[3];
  const float* term_Wih     = (const float*)d_in[4];
  const float* term_Whh     = (const float*)d_in[5];
  const float* term_bih     = (const float*)d_in[6];
  const float* term_bhh     = (const float*)d_in[7];
  const float* doc_emb      = (const float*)d_in[8];
  const float* docc_Wih     = (const float*)d_in[9];
  const float* docc_Whh     = (const float*)d_in[10];
  const float* docc_bih     = (const float*)d_in[11];
  const float* docc_bhh     = (const float*)d_in[12];
  const float* doca_Wih     = (const float*)d_in[13];
  const float* doca_Whh     = (const float*)d_in[14];
  const float* doca_bih     = (const float*)d_in[15];
  const float* doca_bhh     = (const float*)d_in[16];
  const float* pos_table    = (const float*)d_in[17];
  const float* posd_W       = (const float*)d_in[18];
  const float* posd_b       = (const float*)d_in[19];
  const float* qry_Wih      = (const float*)d_in[20];
  /* qry_Whh (d_in[21]) unused: h0 == 0 */
  const float* qry_bih      = (const float*)d_in[22];
  const float* qry_bhh      = (const float*)d_in[23];

  char* ws = (char*)d_ws;
  size_t off = 0;
  auto alloc = [&](size_t bytes) -> char* {
    char* p = ws + off;
    off += (bytes + 255) & ~(size_t)255;
    return p;
  };

  const size_t WMAT = (size_t)G3 * H;           // 196608 elems per weight matrix
  bf16*  WB     = (bf16*)alloc(7 * WMAT * sizeof(bf16));
  bf16*  Xq     = (bf16*)alloc((size_t)LQ * QB * H * sizeof(bf16));
  bf16*  Xp     = (bf16*)alloc((size_t)LD * DPAD * H * sizeof(bf16));
  bf16*  Xn     = (bf16*)alloc((size_t)LD * DPAD * H * sizeof(bf16));
  float* Gi_q   = (float*)alloc((size_t)LQ * QB * G3 * sizeof(float));
  float* Gi_pc  = (float*)alloc((size_t)LD * DPAD * G3 * sizeof(float));
  float* Gi_pa  = (float*)alloc((size_t)LD * DPAD * G3 * sizeof(float));
  float* Gi_nc  = (float*)alloc((size_t)LD * DPAD * G3 * sizeof(float));
  float* Gi_na  = (float*)alloc((size_t)LD * DPAD * G3 * sizeof(float));
  float* hq     = (float*)alloc((size_t)QB * H * sizeof(float));
  bf16*  hqB    = (bf16*)alloc((size_t)QB * H * sizeof(bf16));
  float* hc_p   = (float*)alloc((size_t)DPAD * H * sizeof(float));
  float* ha_p   = (float*)alloc((size_t)DPAD * H * sizeof(float));
  float* hc_n   = (float*)alloc((size_t)DPAD * H * sizeof(float));
  float* ha_n   = (float*)alloc((size_t)DPAD * H * sizeof(float));
  float* cPd    = (float*)alloc((size_t)DPAD * H * sizeof(float));
  float* cNd    = (float*)alloc((size_t)DPAD * H * sizeof(float));
  float* qaug   = (float*)alloc((size_t)QB * H * sizeof(float));
  bf16*  qaugB  = (bf16*)alloc((size_t)QB * H * sizeof(bf16));
  float* Gf     = (float*)alloc((size_t)QB * G3 * sizeof(float));

  // bf16 weight slots: 0 term_Wih, 1 term_Whh, 2 docc_Wih, 3 docc_Whh,
  //                    4 doca_Wih, 5 doca_Whh, 6 qry_Wih
  bf16* wTih = WB + 0 * WMAT; bf16* wThh = WB + 1 * WMAT;
  bf16* wCih = WB + 2 * WMAT; bf16* wChh = WB + 3 * WMAT;
  bf16* wAih = WB + 4 * WMAT; bf16* wAhh = WB + 5 * WMAT;
  bf16* wQih = WB + 6 * WMAT;

  // 1) fp32 -> bf16 weight conversion
  convert_weights<<<dim3(1344), dim3(256), 0, stream>>>(
      term_Wih, term_Whh, docc_Wih, docc_Whh, doca_Wih, doca_Whh, qry_Wih, WB);

  // 2) embedding gathers (time-major bf16)
  gather_embed<<<dim3((LQ * QB * H) / 256), dim3(256), 0, stream>>>(
      query_tokens, term_emb, Xq, LQ, QB, QB);
  gather_embed<<<dim3((LD * DPAD * H) / 256), dim3(256), 0, stream>>>(
      pos_tokens, doc_emb, Xp, LD, DB, DPAD);
  gather_embed<<<dim3((LD * DPAD * H) / 256), dim3(256), 0, stream>>>(
      neg_tokens, doc_emb, Xn, LD, DB, DPAD);

  // 3) precompute input-side gates gi = X @ Wih^T + bih for all timesteps (WMMA)
  wmma_gemm_bias<<<dim3(12, (LQ * QB) / 16), dim3(32), 0, stream>>>(Xq, wTih, term_bih, Gi_q);
  wmma_gemm_bias<<<dim3(12, (LD * DPAD) / 16), dim3(32), 0, stream>>>(Xp, wCih, docc_bih, Gi_pc);
  wmma_gemm_bias<<<dim3(12, (LD * DPAD) / 16), dim3(32), 0, stream>>>(Xp, wAih, doca_bih, Gi_pa);
  wmma_gemm_bias<<<dim3(12, (LD * DPAD) / 16), dim3(32), 0, stream>>>(Xn, wCih, docc_bih, Gi_nc);
  wmma_gemm_bias<<<dim3(12, (LD * DPAD) / 16), dim3(32), 0, stream>>>(Xn, wAih, doca_bih, Gi_na);

  // 4) persistent recurrent scans (h in LDS, Whh streamed from L2, WMMA gates)
  gru_scan_query<<<dim3(1), dim3(512), 0, stream>>>(Gi_q, wThh, term_bhh, hq, hqB);
  gru_scan_docs<<<dim3(2), dim3(256), 0, stream>>>(Gi_pc, Gi_pa, Gi_nc, Gi_na,
                                                   wChh, wAhh, docc_bhh, doca_bhh,
                                                   hc_p, ha_p, hc_n, ha_n);

  // 5) position dense on content vectors (pos + neg concurrently)
  pos_dense<<<dim3(DB, 2), dim3(256), 0, stream>>>(hc_p, hc_n, pos_table,
                                                   posd_W, posd_b, cPd, cNd);

  // 6) attention memory read + combine
  attend_combine<<<dim3(QB), dim3(256), 0, stream>>>(hq, ha_p, cPd, ha_n, cNd, qaug, qaugB);

  // 7) final GRU step (h0 = 0): one WMMA GEMM + gate elementwise
  wmma_gemm_bias<<<dim3(12, QB / 16), dim3(32), 0, stream>>>(qaugB, wQih, qry_bih, Gf);
  final_gate<<<dim3(QB), dim3(256), 0, stream>>>(Gf, qry_bhh, (float*)d_out);
}
